// HeterogeneousMoE_33749853012619
// MI455X (gfx1250) — matrix operand order repaired and checked by reading, not compiled, not run on hardware
//
#include <hip/hip_runtime.h>
#include <math.h>

// ---------------------------------------------------------------------------
// HeterogeneousMoE for MI455X (gfx1250, wave32, WMMA).
// Big GEMMs -> v_wmma_f32_16x16x32_bf16 (f32 accum), fragments fetched as
// contiguous b128 chunks: B pre-packed in fragment order in global memory,
// A staged into LDS already in fragment order (1 ds_store_b128 per thread,
// 2 ds_load_b128 per fragment). Expert outputs fused into d_out with the
// top-2 routing weights (out = residual + sum_e w[b,t,e]*expert_e).
// ---------------------------------------------------------------------------

typedef __bf16 bf16;
typedef __attribute__((ext_vector_type(16))) __bf16 v16bf;
typedef __attribute__((ext_vector_type(8)))  float  v8f;

#define B_      4
#define T_      4096
#define DM      512
#define E_      8
#define DIN_    1024
#define DSTATE_ 16
#define DTRANK_ 32
#define XPO_    64            // DTRANK + 2*DSTATE
#define NTOK    (B_*T_)       // 16384
#define NW_     31            // (T - 256)/128 + 1
#define WIN_    256
#define HOP_    128

__device__ __forceinline__ float gelu_f(float x) {
  return 0.5f * x * (1.0f + erff(x * 0.70710678118654752f));
}
__device__ __forceinline__ float silu_f(float x) {
  return x / (1.0f + __expf(-x));
}
__device__ __forceinline__ float softplus_f(float x) {
  return (x > 20.0f) ? x : log1pf(__expf(x));
}

// ---------------------------------------------------------------------------
// 0) zero small accumulators: acc[0..3]=ent sums, acc[4..11]=tpe, acc[12..19]=avg_probs
__global__ void k_zero_accum(float* acc) {
  if (threadIdx.x < 20) acc[threadIdx.x] = 0.0f;
}

// ---------------------------------------------------------------------------
// 1) LayerNorm + write residual into d_out + xn (f32 and bf16)
__global__ __launch_bounds__(256) void k_layernorm(
    const float* __restrict__ x, const float* __restrict__ g,
    const float* __restrict__ be, float* __restrict__ xn,
    bf16* __restrict__ xnb, float* __restrict__ out) {
  __shared__ float red[256];
  const int t = blockIdx.x, tid = threadIdx.x;
  const size_t base = (size_t)t * DM;
  const float a0 = x[base + tid];
  const float a1 = x[base + tid + 256];
  red[tid] = a0 + a1; __syncthreads();
  for (int s = 128; s > 0; s >>= 1) { if (tid < s) red[tid] += red[tid + s]; __syncthreads(); }
  const float mean = red[0] * (1.0f / 512.0f);
  __syncthreads();
  const float d0 = a0 - mean, d1 = a1 - mean;
  red[tid] = d0 * d0 + d1 * d1; __syncthreads();
  for (int s = 128; s > 0; s >>= 1) { if (tid < s) red[tid] += red[tid + s]; __syncthreads(); }
  const float rs = rsqrtf(red[0] * (1.0f / 512.0f) + 1e-5f);
  const float y0 = d0 * rs * g[tid] + be[tid];
  const float y1 = d1 * rs * g[tid + 256] + be[tid + 256];
  xn[base + tid] = y0;          xn[base + tid + 256] = y1;
  xnb[base + tid] = (bf16)y0;   xnb[base + tid + 256] = (bf16)y1;
  out[base + tid] = a0;         out[base + tid + 256] = a1;   // residual init
}

// ---------------------------------------------------------------------------
// 2) Spectral entropy: one block per (batch, window, channel); naive 256-pt rDFT
__global__ __launch_bounds__(256) void k_entropy(
    const float* __restrict__ xn, float* __restrict__ acc) {
  __shared__ float win[WIN_];
  __shared__ float red[256];
  const int bid = blockIdx.x;
  const int c  = bid & (DM - 1);
  const int wi = (bid >> 9) % NW_;
  const int b  = bid / (DM * NW_);
  const int tid = threadIdx.x;
  win[tid] = xn[((size_t)(b * T_ + wi * HOP_ + tid)) * DM + c];
  __syncthreads();
  const int f = tid;
  float spec = 0.0f;
  if (f < 129) {
    const float c0 = -6.283185307179586f * (float)f * (1.0f / 256.0f);
    float re = 0.0f, im = 0.0f;
    for (int j = 0; j < WIN_; ++j) {
      float si, co;
      __sincosf(c0 * (float)j, &si, &co);
      re += win[j] * co;
      im += win[j] * si;
    }
    spec = sqrtf(re * re + im * im) + 1e-10f;
  }
  red[tid] = spec; __syncthreads();
  for (int s = 128; s > 0; s >>= 1) { if (tid < s) red[tid] += red[tid + s]; __syncthreads(); }
  const float tot = red[0];
  __syncthreads();
  float contrib = 0.0f;
  if (f < 129) {
    const float p = spec / tot;
    contrib = -p * __logf(p + 1e-10f);
  }
  red[tid] = contrib; __syncthreads();
  for (int s = 128; s > 0; s >>= 1) { if (tid < s) red[tid] += red[tid + s]; __syncthreads(); }
  if (tid == 0) atomicAdd(&acc[b], red[0] * (1.0f / __logf(129.0f)));
}

// ---------------------------------------------------------------------------
// 3) Gating: logits, temperature, top-2 softmax weights, aux accumulators
__global__ __launch_bounds__(256) void k_gating(
    const float* __restrict__ xn, const float* __restrict__ gw,
    const float* __restrict__ entw, const float* __restrict__ entb,
    const float* __restrict__ temp, float* __restrict__ acc,
    float* __restrict__ wfull) {
  __shared__ float s_tpe[E_], s_avg[E_];
  const int tid = threadIdx.x;
  if (tid < E_) { s_tpe[tid] = 0.0f; s_avg[tid] = 0.0f; }
  __syncthreads();

  const int t = blockIdx.x * 256 + tid;
  float lg[E_];
  #pragma unroll
  for (int e = 0; e < E_; ++e) lg[e] = 0.0f;
  const float* xr = xn + (size_t)t * DM;
  for (int k = 0; k < DM; ++k) {
    const float xv = xr[k];
    #pragma unroll
    for (int e = 0; e < E_; ++e) lg[e] += xv * gw[k * E_ + e];
  }
  const int b = t >> 12;
  const float entv = acc[b] * (1.0f / (float)(NW_ * DM));
  const float invT = 1.0f / (fabsf(temp[0]) + 1e-6f);
  #pragma unroll
  for (int e = 0; e < E_; ++e) lg[e] = (lg[e] + entv * entw[e] + entb[e]) * invT;

  // full softmax for avg_probs
  float mx = lg[0];
  #pragma unroll
  for (int e = 1; e < E_; ++e) mx = fmaxf(mx, lg[e]);
  float se = 0.0f;
  float pe[E_];
  #pragma unroll
  for (int e = 0; e < E_; ++e) { pe[e] = __expf(lg[e] - mx); se += pe[e]; }
  const float inv_se = 1.0f / se;
  #pragma unroll
  for (int e = 0; e < E_; ++e) atomicAdd(&s_avg[e], pe[e] * inv_se);

  // top-2 (first-occurrence on ties, like jax top_k)
  int i0 = 0; float v0 = lg[0];
  #pragma unroll
  for (int e = 1; e < E_; ++e) if (lg[e] > v0) { v0 = lg[e]; i0 = e; }
  int i1 = (i0 == 0) ? 1 : 0; float v1 = lg[i1];
  #pragma unroll
  for (int e = 0; e < E_; ++e) if (e != i0 && lg[e] > v1) { v1 = lg[e]; i1 = e; }
  const float e1 = __expf(v1 - v0);
  const float r0 = 1.0f / (1.0f + e1);
  const float r1 = e1 * r0;
  #pragma unroll
  for (int e = 0; e < E_; ++e) wfull[(size_t)t * E_ + e] = 0.0f;
  wfull[(size_t)t * E_ + i0] = r0;
  wfull[(size_t)t * E_ + i1] = r1;
  atomicAdd(&s_tpe[i0], 1.0f);
  atomicAdd(&s_tpe[i1], 1.0f);
  __syncthreads();
  if (tid < E_) {
    atomicAdd(&acc[4 + tid],  s_tpe[tid]);
    atomicAdd(&acc[12 + tid], s_avg[tid]);
  }
}

// ---------------------------------------------------------------------------
// 4) Pack f32 weights (K x N, row-major) into bf16 WMMA-B fragment order:
//    out[ct][ks][nt][lane][pos], ct=n/64, ks=k/32, nt=(n%64)/16,
//    lane=(n&15)+(((k&31)>>4)<<4), pos=k&15  -> each lane's fragment is a
//    contiguous 32B run, coalesced across lanes.
__global__ void k_pack_w(const float* __restrict__ w, bf16* __restrict__ out,
                         int K, int N) {
  const int idx = blockIdx.x * blockDim.x + threadIdx.x;
  if (idx >= K * N) return;
  const int k = idx / N;
  const int n = idx - k * N;
  const int ct = n >> 6, nn = n & 63;
  const int ks = k >> 5, kk = k & 31;
  const int nt = nn >> 4;
  const int lane = (nn & 15) + ((kk >> 4) << 4);
  const int pos = kk & 15;
  const size_t pidx =
      ((((size_t)ct * (K >> 5) + ks) * 4 + nt) * 32 + lane) * 16 + pos;
  out[pidx] = (bf16)w[idx];
}

// ---------------------------------------------------------------------------
// 5) WMMA GEMM: C(MxN) = epilogue(A_bf16(MxK,row-major) @ Wpacked + bias)
//    64x64 block tile, K-step 32, 256 threads = 8 waves, 2 wmma per wave/step.
//    A staged into LDS in fragment order; B read directly from packed global.
//    EPI: 0 = GELU store f32, 1 = store f32, 2 = MoE-weighted accumulate.
template <int EPI>
__global__ __launch_bounds__(256) void k_gemm_wmma(
    const bf16* __restrict__ A, const bf16* __restrict__ Bp,
    const float* __restrict__ bias, float* __restrict__ C,
    int M, int N, int K, int ldc,
    const float* __restrict__ wfull, int expert) {
  // A tile in fragment order: [mt][lane][pos] = 4*32*16 bf16 = 4 KB
  __shared__ __align__(16) bf16 sA[4 * 32 * 16];
  const int tid  = threadIdx.x;
  const int lane = tid & 31;
  const int w    = tid >> 5;
  const int mt     = w & 3;              // 16-row subtile within 64-row block
  const int ntBase = (w >> 2) * 2;       // first of two 16-col subtiles
  const int rowBase = blockIdx.y * 64;
  const int colBase = blockIdx.x * 64;
  const int ksteps  = K >> 5;

  // staging: thread loads 8 contiguous bf16 of A row (rowBase+tid/4),
  // k0=(tid&3)*8 within the 32-wide K step; lands contiguously in one
  // lane's fragment (K-grouping {0-15}/{16-31}, matching packed B).
  const int aSrcRow = rowBase + (tid >> 2);
  const int aSrcCol = (tid & 3) * 8;
  const int j       = tid & 3;
  const int aDst    = ((tid >> 6) * 32 + ((tid >> 2) & 15) + ((j >> 1) << 4)) * 16
                      + ((j & 1) << 3);

  union { v8f v; float e[8]; } acc0, acc1;
  #pragma unroll
  for (int i = 0; i < 8; ++i) { acc0.e[i] = 0.0f; acc1.e[i] = 0.0f; }

  const size_t bTileBase = ((size_t)blockIdx.x * ksteps * 4 + ntBase) * 32 * 16;

  for (int ks = 0; ks < ksteps; ++ks) {
    *(uint4*)&sA[aDst] =
        *(const uint4*)&A[(size_t)aSrcRow * K + (ks << 5) + aSrcCol];
    if (ks + 1 < ksteps) {
      __builtin_prefetch(&A[(size_t)aSrcRow * K + ((ks + 1) << 5) + aSrcCol], 0, 1);
      __builtin_prefetch(&Bp[bTileBase + (size_t)(ks + 1) * 4 * 32 * 16 + lane * 16], 0, 1);
    }
    __syncthreads();

    union { v16bf v; uint4 q[2]; } fa, fb0, fb1;
    const bf16* ap = &sA[(mt * 32 + lane) * 16];
    fa.q[0] = *(const uint4*)ap;
    fa.q[1] = *(const uint4*)(ap + 8);

    const bf16* bp = Bp + bTileBase + (size_t)ks * 4 * 32 * 16 + lane * 16;
    fb0.q[0] = *(const uint4*)bp;
    fb0.q[1] = *(const uint4*)(bp + 8);
    fb1.q[0] = *(const uint4*)(bp + 512);       // next nt subtile (+32*16)
    fb1.q[1] = *(const uint4*)(bp + 520);

    acc0.v = __builtin_amdgcn_wmma_f32_16x16x32_bf16(
        false, fa.v, false, fb0.v, (short)0, acc0.v, false, false);
    acc1.v = __builtin_amdgcn_wmma_f32_16x16x32_bf16(
        false, fa.v, false, fb1.v, (short)0, acc1.v, false, false);
    __syncthreads();
  }

  // C/D layout: VGPR i -> lanes 0-15 M=i, lanes 16-31 M=8+i
  #pragma unroll
  for (int i = 0; i < 8; ++i) {
    const int m   = mt * 16 + ((lane >> 4) << 3) + i;
    const int row = rowBase + m;
    #pragma unroll
    for (int half = 0; half < 2; ++half) {
      const int col = colBase + (ntBase + half) * 16 + (lane & 15);
      float v = half ? acc1.e[i] : acc0.e[i];
      if (bias) v += bias[col];
      if (EPI == 0) {
        C[(size_t)row * ldc + col] = gelu_f(v);
      } else if (EPI == 1) {
        C[(size_t)row * ldc + col] = v;
      } else {
        C[(size_t)row * ldc + col] += wfull[(size_t)row * E_ + expert] * v;
      }
    }
  }
}

// ---------------------------------------------------------------------------
// 6) conv expert: causal depthwise conv K=7 + bias, then GELU, emit bf16
__global__ void k_dwconv7_gelu(const float* __restrict__ h, const float* __restrict__ kw,
                               const float* __restrict__ kb, bf16* __restrict__ out) {
  const int idx = blockIdx.x * blockDim.x + threadIdx.x;   // tok*1024 + c
  const int c   = idx & (DIN_ - 1);
  const int tok = idx >> 10;
  const int t   = tok & (T_ - 1);
  const int b   = tok >> 12;
  float acc = kb[c];
  #pragma unroll
  for (int jj = 0; jj < 7; ++jj) {
    const int tt = t - 6 + jj;
    if (tt >= 0) acc += h[(size_t)(b * T_ + tt) * DIN_ + c] * kw[c * 7 + jj];
  }
  out[idx] = (bf16)gelu_f(acc);
}

// 7) mamba: causal depthwise conv K=4 + bias on u (stride-2048 view), then SiLU
__global__ void k_dwconv4_silu(const float* __restrict__ uz, const float* __restrict__ cw,
                               const float* __restrict__ cb, float* __restrict__ uconv) {
  const int idx = blockIdx.x * blockDim.x + threadIdx.x;
  const int c   = idx & (DIN_ - 1);
  const int tok = idx >> 10;
  const int t   = tok & (T_ - 1);
  const int b   = tok >> 12;
  float acc = cb[c];
  #pragma unroll
  for (int jj = 0; jj < 4; ++jj) {
    const int tt = t - 3 + jj;
    if (tt >= 0) acc += uz[(size_t)(b * T_ + tt) * (2 * DIN_) + c] * cw[c * 4 + jj];
  }
  uconv[idx] = silu_f(acc);
}

// 8) dbc = uconv @ x_proj  (K=1024, N=64; small -> VALU)
__global__ void k_dbc(const float* __restrict__ uconv, const float* __restrict__ xp,
                      float* __restrict__ dbc) {
  const int idx = blockIdx.x * blockDim.x + threadIdx.x;  // tok*64 + o
  const int o   = idx & (XPO_ - 1);
  const int tok = idx >> 6;
  const float* ur = uconv + (size_t)tok * DIN_;
  float acc = 0.0f;
  for (int k = 0; k < DIN_; ++k) acc += ur[k] * xp[k * XPO_ + o];
  dbc[idx] = acc;
}

// 9) selective scan: one thread per (batch, channel); delta recomputed on the fly
__global__ __launch_bounds__(256) void k_scan(
    const float* __restrict__ uconv, const float* __restrict__ uz,
    const float* __restrict__ dbc, const float* __restrict__ dtw,
    const float* __restrict__ dtb, const float* __restrict__ alog,
    const float* __restrict__ Dp, bf16* __restrict__ yact) {
  const int idx = blockIdx.x * blockDim.x + threadIdx.x;  // b*1024 + c
  const int c = idx & (DIN_ - 1);
  const int b = idx >> 10;
  float dtwc[DTRANK_];
  #pragma unroll
  for (int r = 0; r < DTRANK_; ++r) dtwc[r] = dtw[r * DIN_ + c];
  float As[DSTATE_], hst[DSTATE_];
  #pragma unroll
  for (int s = 0; s < DSTATE_; ++s) { As[s] = -__expf(alog[c * DSTATE_ + s]); hst[s] = 0.0f; }
  const float dtbc = dtb[c], Dc = Dp[c];
  for (int t = 0; t < T_; ++t) {
    const size_t tok = (size_t)b * T_ + t;
    const float* row = dbc + tok * XPO_;
    float d = dtbc;
    #pragma unroll
    for (int r = 0; r < DTRANK_; ++r) d += row[r] * dtwc[r];
    d = softplus_f(d);
    const float ut = uconv[tok * DIN_ + c];
    float y = ut * Dc;
    #pragma unroll
    for (int s = 0; s < DSTATE_; ++s) {
      hst[s] = __expf(d * As[s]) * hst[s] + d * row[DTRANK_ + s] * ut;
      y += hst[s] * row[DTRANK_ + DSTATE_ + s];
    }
    const float z = uz[tok * (2 * DIN_) + DIN_ + c];
    y *= silu_f(z);
    yact[tok * DIN_ + c] = (bf16)y;
  }
}

// 10) aux = E * sum_e (tpe[e]/NTOK)*(avgp[e]/NTOK)
__global__ void k_finalize(const float* __restrict__ acc, float* __restrict__ out) {
  float aux = 0.0f;
  #pragma unroll
  for (int e = 0; e < E_; ++e)
    aux += (acc[4 + e] * (1.0f / NTOK)) * (acc[12 + e] * (1.0f / NTOK));
  out[(size_t)NTOK * DM] = (float)E_ * aux;
}

// ---------------------------------------------------------------------------
extern "C" void kernel_launch(void* const* d_in, const int* in_sizes, int n_in,
                              void* d_out, int out_size, void* d_ws, size_t ws_size,
                              hipStream_t stream) {
  (void)in_sizes; (void)n_in; (void)out_size; (void)ws_size;
  const float* x        = (const float*)d_in[0];
  const float* ln_g     = (const float*)d_in[1];
  const float* ln_b     = (const float*)d_in[2];
  const float* gate_w   = (const float*)d_in[3];
  const float* ent_w    = (const float*)d_in[4];
  const float* ent_b    = (const float*)d_in[5];
  const float* temp     = (const float*)d_in[6];
  const float* cin_w    = (const float*)d_in[7];
  const float* cin_b    = (const float*)d_in[8];
  const float* conv_k   = (const float*)d_in[9];
  const float* conv_kb  = (const float*)d_in[10];
  const float* cout_w   = (const float*)d_in[11];
  const float* cout_b   = (const float*)d_in[12];
  const float* inproj   = (const float*)d_in[13];
  const float* mconv_w  = (const float*)d_in[14];
  const float* mconv_b  = (const float*)d_in[15];
  const float* xproj    = (const float*)d_in[16];
  const float* dt_w     = (const float*)d_in[17];
  const float* dt_b     = (const float*)d_in[18];
  const float* A_log    = (const float*)d_in[19];
  const float* Dp       = (const float*)d_in[20];
  const float* outproj  = (const float*)d_in[21];
  float* out = (float*)d_out;
  char*  ws  = (char*)d_ws;

  // workspace sub-allocation (256B aligned); ~312 MB total
  size_t off = 0;
  auto walloc = [&](size_t bytes) { size_t o = off; off = (off + bytes + 255) & ~(size_t)255; return o; };
  float* xn      = (float*)(ws + walloc((size_t)NTOK * DM * 4));          // 33.5 MB
  bf16*  xnb     = (bf16*) (ws + walloc((size_t)NTOK * DM * 2));          // 16.8 MB
  float* wfull   = (float*)(ws + walloc((size_t)NTOK * E_ * 4));          // 0.5 MB
  float* acc     = (float*)(ws + walloc(1024));
  bf16*  wbf_cin = (bf16*) (ws + walloc((size_t)4 * DM * DIN_ * 2));      // 4.2 MB (packed)
  bf16*  wbf_cout= (bf16*) (ws + walloc((size_t)4 * DIN_ * DM * 2));      // 4.2 MB (packed)
  bf16*  wbf_inp = (bf16*) (ws + walloc((size_t)4 * DM * 2 * DIN_ * 2));  // 8.4 MB (packed)
  bf16*  wbf_outp= (bf16*) (ws + walloc((size_t)4 * DIN_ * DM * 2));      // 4.2 MB (packed)
  float* bufUZ   = (float*)(ws + walloc((size_t)NTOK * 2 * DIN_ * 4));    // 134 MB
  float* bufB    = (float*)(ws + walloc((size_t)NTOK * DIN_ * 4));        // 67 MB
  bf16*  actbf   = (bf16*) (ws + walloc((size_t)NTOK * DIN_ * 2));        // 33.5 MB
  float* dbc     = (float*)(ws + walloc((size_t)NTOK * XPO_ * 4));        // 4.2 MB

  k_zero_accum<<<1, 32, 0, stream>>>(acc);
  k_layernorm<<<NTOK, 256, 0, stream>>>(x, ln_g, ln_b, xn, xnb, out);
  k_entropy<<<B_ * NW_ * DM, 256, 0, stream>>>(xn, acc);
  k_gating<<<NTOK / 256, 256, 0, stream>>>(xn, gate_w, ent_w, ent_b, temp, acc, wfull);

  // pack weights into bf16 WMMA-fragment order (per expert slice)
  for (int e = 0; e < 4; ++e) {
    int n;
    n = DM * DIN_;
    k_pack_w<<<(n + 255) / 256, 256, 0, stream>>>(
        cin_w + (size_t)e * n, wbf_cin + (size_t)e * n, DM, DIN_);
    n = DIN_ * DM;
    k_pack_w<<<(n + 255) / 256, 256, 0, stream>>>(
        cout_w + (size_t)e * n, wbf_cout + (size_t)e * n, DIN_, DM);
    n = DM * 2 * DIN_;
    k_pack_w<<<(n + 255) / 256, 256, 0, stream>>>(
        inproj + (size_t)e * n, wbf_inp + (size_t)e * n, DM, 2 * DIN_);
    n = DIN_ * DM;
    k_pack_w<<<(n + 255) / 256, 256, 0, stream>>>(
        outproj + (size_t)e * n, wbf_outp + (size_t)e * n, DIN_, DM);
  }

  const dim3 blk(256);
  // --- conv experts (e = 0..3) ---
  for (int e = 0; e < 4; ++e) {
    k_gemm_wmma<0><<<dim3(DIN_ / 64, NTOK / 64), blk, 0, stream>>>(
        xnb, wbf_cin + (size_t)e * DM * DIN_, cin_b + (size_t)e * DIN_,
        bufB, NTOK, DIN_, DM, DIN_, nullptr, 0);
    k_dwconv7_gelu<<<(NTOK * DIN_) / 256, blk, 0, stream>>>(
        bufB, conv_k + (size_t)e * DIN_ * 7, conv_kb + (size_t)e * DIN_, actbf);
    k_gemm_wmma<2><<<dim3(DM / 64, NTOK / 64), blk, 0, stream>>>(
        actbf, wbf_cout + (size_t)e * DIN_ * DM, cout_b + (size_t)e * DM,
        out, NTOK, DM, DIN_, DM, wfull, e);
  }

  // --- mamba experts (e = 4..7) ---
  for (int e = 0; e < 4; ++e) {
    k_gemm_wmma<1><<<dim3(2 * DIN_ / 64, NTOK / 64), blk, 0, stream>>>(
        xnb, wbf_inp + (size_t)e * DM * 2 * DIN_, nullptr,
        bufUZ, NTOK, 2 * DIN_, DM, 2 * DIN_, nullptr, 0);
    k_dwconv4_silu<<<(NTOK * DIN_) / 256, blk, 0, stream>>>(
        bufUZ, mconv_w + (size_t)e * DIN_ * 4, mconv_b + (size_t)e * DIN_, bufB);
    k_dbc<<<(NTOK * XPO_) / 256, blk, 0, stream>>>(
        bufB, xproj + (size_t)e * DIN_ * XPO_, dbc);
    k_scan<<<(B_ * DIN_) / 256, blk, 0, stream>>>(
        bufB, bufUZ, dbc, dt_w + (size_t)e * DTRANK_ * DIN_, dt_b + (size_t)e * DIN_,
        A_log + (size_t)e * DIN_ * DSTATE_, Dp + (size_t)e * DIN_, actbf);
    k_gemm_wmma<2><<<dim3(DM / 64, NTOK / 64), blk, 0, stream>>>(
        actbf, wbf_outp + (size_t)e * DIN_ * DM, nullptr,
        out, NTOK, DM, DIN_, DM, wfull, 4 + e);
  }

  k_finalize<<<1, 1, 0, stream>>>(acc, out);
}
